// GCN2_3418793968077
// MI455X (gfx1250) — matrix operand order repaired and checked by reading, not compiled, not run on hardware
//
#include <hip/hip_runtime.h>
#include <hip/hip_bf16.h>
#include <stdint.h>

// ---------------------------------------------------------------------------
// GCN (2-layer) for MI455X / gfx1250, wave32 + WMMA bf16 split-precision GEMM.
// W is pre-packed into the ISA 7.12.2 B-fragment layout (bf16 hi + lo).
// One wave computes a 16 x (16*NT) output strip -> 3*NT WMMAs per K-step.
// ---------------------------------------------------------------------------

#define NFEAT 512
#define NHID  128
#define NCLASS 16

typedef __attribute__((ext_vector_type(16))) __bf16 v16bf;
typedef __attribute__((ext_vector_type(8)))  float  v8f;

// ---------------------------------------------------------------------------
// Pack W [K x Nc] (row-major fp32) into B-fragment layout:
//   frag index f = (tile_n * (K/32) + ks) * 32 + lane ; 16 bf16 per fragment.
//   lane: col = tile_n*16 + (lane&15), K = ks*32 + 8*(lane>>4) + (i<8 ? i : i+8).
// hi = rn(bf16), lo = rn(bf16(v - hi))  (split-precision pair).
// ---------------------------------------------------------------------------
template <int K>
__global__ void pack_w_kernel(const float* __restrict__ W, int Nc,
                              __bf16* __restrict__ whi,
                              __bf16* __restrict__ wlo) {
    int t = blockIdx.x * blockDim.x + threadIdx.x;   // one fragment per thread
    int total = (Nc / 16) * (K / 32) * 32;
    if (t >= total) return;
    int lane = t & 31;
    int ks   = (t >> 5) % (K / 32);
    int tn   = (t >> 5) / (K / 32);
    int half = lane >> 4;
    int l15  = lane & 15;
    const float* __restrict__ bcol = W + tn * 16 + l15;
    int k0 = ks * 32 + 8 * half;
#pragma unroll
    for (int i = 0; i < 16; ++i) {
        float v = bcol[(size_t)(k0 + (i < 8 ? i : i + 8)) * Nc];
        __bf16 h = (__bf16)v;
        whi[(size_t)t * 16 + i] = h;
        wlo[(size_t)t * 16 + i] = (__bf16)(v - (float)h);
    }
}

// ---------------------------------------------------------------------------
// C[M x 16*NT] = op(A)[M x K] @ B ;  B pre-packed (whi/wlo fragments).
// op(A) = relu(A) if RELU.  One wave -> one 16-row strip across all NT tiles.
// 4 waves per block. Requires M % 16 == 0, K % 32 == 0.
// ---------------------------------------------------------------------------
template <int K, int NT, bool RELU>
__global__ void wmma_gemm_kernel(const float* __restrict__ A,
                                 const __bf16* __restrict__ whi,
                                 const __bf16* __restrict__ wlo,
                                 float* __restrict__ C, int M) {
    constexpr int Nc = NT * 16;
    constexpr int KS = K / 32;
    const int wave   = threadIdx.x >> 5;
    const int lane   = threadIdx.x & 31;
    const int half   = lane >> 4;
    const int l15    = lane & 15;
    const int mTiles = M >> 4;
    const int tile_m = blockIdx.y * 4 + wave;
    if (tile_m >= mTiles) return;        // wave-uniform: EXEC stays all-ones

    const float* __restrict__ arow = A + (size_t)(tile_m * 16 + l15) * K;
    const v16bf* __restrict__ bhiF = (const v16bf*)whi;
    const v16bf* __restrict__ bloF = (const v16bf*)wlo;

    v8f acc[NT];
#pragma unroll
    for (int nt = 0; nt < NT; ++nt) acc[nt] = (v8f){};

    for (int ks = 0; ks < KS; ++ks) {
        // ---- A fragment (16x32 bf16 layout): 4 x float4 contiguous loads ---
        const float* aptr = arow + ks * 32 + 8 * half;
        float4 p0 = *(const float4*)(aptr + 0);
        float4 p1 = *(const float4*)(aptr + 4);
        float4 q0 = *(const float4*)(aptr + 16);
        float4 q1 = *(const float4*)(aptr + 20);
        float av[16] = {p0.x, p0.y, p0.z, p0.w, p1.x, p1.y, p1.z, p1.w,
                        q0.x, q0.y, q0.z, q0.w, q1.x, q1.y, q1.z, q1.w};
        v16bf a_hi, a_lo;
#pragma unroll
        for (int i = 0; i < 16; ++i) {
            float v = av[i];
            if (RELU) v = fmaxf(v, 0.0f);
            __bf16 h = (__bf16)v;
            a_hi[i] = h;
            a_lo[i] = (__bf16)(v - (float)h);
        }

        // ---- NT N-tiles reuse the A fragment: 3*NT WMMAs per K-step --------
#pragma unroll
        for (int nt = 0; nt < NT; ++nt) {
            size_t f = (size_t)(nt * KS + ks) * 32 + lane;
            v16bf b_hi = bhiF[f];
            v16bf b_lo = bloF[f];
            acc[nt] = __builtin_amdgcn_wmma_f32_16x16x32_bf16(
                false, a_hi, false, b_hi, (short)0, acc[nt], false, false);
            acc[nt] = __builtin_amdgcn_wmma_f32_16x16x32_bf16(
                false, a_hi, false, b_lo, (short)0, acc[nt], false, false);
            acc[nt] = __builtin_amdgcn_wmma_f32_16x16x32_bf16(
                false, a_lo, false, b_hi, (short)0, acc[nt], false, false);
        }
    }

    // C/D layout: VGPR j -> M = j + 8*half, N = l15 (per tile)
    float* __restrict__ cbase = C + (size_t)tile_m * 16 * Nc + l15;
#pragma unroll
    for (int nt = 0; nt < NT; ++nt) {
        float* crow = cbase + nt * 16;
#pragma unroll
        for (int j = 0; j < 8; ++j)
            crow[(size_t)(j + 8 * half) * Nc] = acc[nt][j];
    }
}

// ---------------------------------------------------------------------------
// Degree / normalization
// ---------------------------------------------------------------------------
__global__ void init_deg_kernel(float* __restrict__ deg, int n) {
    int i = blockIdx.x * blockDim.x + threadIdx.x;
    if (i < n) deg[i] = 1.0f;            // self-loop contribution
}

__global__ void count_deg_kernel(const int* __restrict__ dst,
                                 float* __restrict__ deg, int e) {
    int i = blockIdx.x * blockDim.x + threadIdx.x;
    if (i < e) atomicAdd(&deg[dst[i]], 1.0f);
}

__global__ void dinv_kernel(const float* __restrict__ deg,
                            float* __restrict__ dinv, int n) {
    int i = blockIdx.x * blockDim.x + threadIdx.x;
    if (i < n) {
        float d = deg[i];
        dinv[i] = (d > 0.0f) ? rsqrtf(d) : 0.0f;
    }
}

// ---------------------------------------------------------------------------
// Aggregation: out = bias + self_loop + sum_edges
// ---------------------------------------------------------------------------
__global__ void agg_init_kernel(const float* __restrict__ h,
                                const float* __restrict__ dinv,
                                const float* __restrict__ bias,
                                float* __restrict__ out, int n) {
    int t = blockIdx.x * blockDim.x + threadIdx.x;          // n * 32 threads
    if (t >= n * (NHID / 4)) return;
    int node = t >> 5;
    int c    = t & 31;
    float s  = dinv[node];
    s = s * s;
    const float4 hv = ((const float4*)h)[t];
    const float4 bv = ((const float4*)bias)[c];
    float4 o;
    o.x = hv.x * s + bv.x;
    o.y = hv.y * s + bv.y;
    o.z = hv.z * s + bv.z;
    o.w = hv.w * s + bv.w;
    ((float4*)out)[t] = o;
}

// Edge scatter: thread = (edge, 4-feature chunk). h/out rows are L2-resident.
__global__ void agg_scatter_kernel(const int* __restrict__ src,
                                   const int* __restrict__ dst,
                                   const float* __restrict__ dinv,
                                   const float* __restrict__ h,
                                   float* __restrict__ out, int e) {
    int t = blockIdx.x * blockDim.x + threadIdx.x;          // e * 32 threads
    if (t >= e * (NHID / 4)) return;
    int edge = t >> 5;
    int c    = t & 31;
    int s    = src[edge];
    int d    = dst[edge];
    float norm = dinv[s] * dinv[d];
    const float4 hv = ((const float4*)(h + (size_t)s * NHID))[c];
    float* o = out + (size_t)d * NHID + c * 4;
    atomicAdd(o + 0, hv.x * norm);
    atomicAdd(o + 1, hv.y * norm);
    atomicAdd(o + 2, hv.z * norm);
    atomicAdd(o + 3, hv.w * norm);
}

// ---------------------------------------------------------------------------
// log_softmax over 16 classes (+ classifier bias), one thread per node.
// ---------------------------------------------------------------------------
__global__ void logsoftmax_kernel(const float* __restrict__ logits,
                                  const float* __restrict__ bc,
                                  float* __restrict__ out, int n) {
    int i = blockIdx.x * blockDim.x + threadIdx.x;
    if (i >= n) return;
    const float* row = logits + (size_t)i * NCLASS;
    float v[NCLASS];
    float mx = -3.402823466e38f;
#pragma unroll
    for (int c = 0; c < NCLASS; ++c) {
        v[c] = row[c] + bc[c];
        mx = fmaxf(mx, v[c]);
    }
    float s = 0.0f;
#pragma unroll
    for (int c = 0; c < NCLASS; ++c) s += __expf(v[c] - mx);
    float lse = mx + __logf(s);
    float* o = out + (size_t)i * NCLASS;
#pragma unroll
    for (int c = 0; c < NCLASS; ++c) o[c] = v[c] - lse;
}

// ---------------------------------------------------------------------------
// Launcher
// ---------------------------------------------------------------------------
extern "C" void kernel_launch(void* const* d_in, const int* in_sizes, int n_in,
                              void* d_out, int out_size, void* d_ws, size_t ws_size,
                              hipStream_t stream) {
    const float* x   = (const float*)d_in[0];
    const int*   ei  = (const int*)d_in[1];   // [2, E] flattened: src row, dst row
    const float* W1  = (const float*)d_in[2];
    const float* b1  = (const float*)d_in[3];
    const float* W2  = (const float*)d_in[4];
    const float* b2  = (const float*)d_in[5];
    const float* Wc  = (const float*)d_in[6];
    const float* bc  = (const float*)d_in[7];
    float*       out = (float*)d_out;

    const int N = in_sizes[0] / NFEAT;        // 50000
    const int E = in_sizes[1] / 2;            // 800000
    const int* e_src = ei;
    const int* e_dst = ei + E;

    // Packed-fragment element counts: (Nc/16)*(K/32)*32 fragments * 16 bf16
    const size_t nW1 = (size_t)(NHID / 16) * (NFEAT / 32) * 32 * 16;   // 65536
    const size_t nW2 = (size_t)(NHID / 16) * (NHID / 32) * 32 * 16;    // 16384
    const size_t nWc = (size_t)(NCLASS / 16) * (NHID / 32) * 32 * 16;  //  2048

    // Workspace layout (256-byte aligned chunks)
    uintptr_t p = ((uintptr_t)d_ws + 255) & ~(uintptr_t)255;
    float* deg  = (float*)p;              p += (size_t)N * 4;           p = (p + 255) & ~(uintptr_t)255;
    float* dinv = (float*)p;              p += (size_t)N * 4;           p = (p + 255) & ~(uintptr_t)255;
    __bf16* w1h = (__bf16*)p;             p += nW1 * 2;
    __bf16* w1l = (__bf16*)p;             p += nW1 * 2;
    __bf16* w2h = (__bf16*)p;             p += nW2 * 2;
    __bf16* w2l = (__bf16*)p;             p += nW2 * 2;
    __bf16* wch = (__bf16*)p;             p += nWc * 2;
    __bf16* wcl = (__bf16*)p;             p += nWc * 2;                 p = (p + 255) & ~(uintptr_t)255;
    float* bufA = (float*)p;              p += (size_t)N * NHID * 4;    p = (p + 255) & ~(uintptr_t)255;
    float* bufB = (float*)p;

    const int BT = 256;
    const int mTiles = N / 16;                                  // 3125
    const dim3 gemmBlk(128);                                    // 4 waves
    const dim3 gemmGrd(1, (mTiles + 3) / 4);

    // --- pack weights into WMMA B-fragment layout ---------------------------
    pack_w_kernel<NFEAT><<<(4096 + BT - 1) / BT, BT, 0, stream>>>(W1, NHID, w1h, w1l);
    pack_w_kernel<NHID><<<(1024 + BT - 1) / BT, BT, 0, stream>>>(W2, NHID, w2h, w2l);
    pack_w_kernel<NHID><<<(128 + BT - 1) / BT, BT, 0, stream>>>(Wc, NCLASS, wch, wcl);

    // --- normalization ------------------------------------------------------
    init_deg_kernel<<<(N + BT - 1) / BT, BT, 0, stream>>>(deg, N);
    count_deg_kernel<<<(E + BT - 1) / BT, BT, 0, stream>>>(e_dst, deg, E);
    dinv_kernel<<<(N + BT - 1) / BT, BT, 0, stream>>>(deg, dinv, N);

    // --- layer 1: h1 = x @ W1 ; out1 = Agg(h1) + b1 -------------------------
    wmma_gemm_kernel<NFEAT, NHID / 16, false><<<gemmGrd, gemmBlk, 0, stream>>>(x, w1h, w1l, bufA, N);
    agg_init_kernel<<<((size_t)N * 32 + BT - 1) / BT, BT, 0, stream>>>(bufA, dinv, b1, bufB, N);
    agg_scatter_kernel<<<((size_t)E * 32 + BT - 1) / BT, BT, 0, stream>>>(
        e_src, e_dst, dinv, bufA, bufB, E);

    // --- layer 2: h2 = relu(out1) @ W2 ; out2 = Agg(h2) + b2 ----------------
    wmma_gemm_kernel<NHID, NHID / 16, true><<<gemmGrd, gemmBlk, 0, stream>>>(bufB, w2h, w2l, bufA, N);
    agg_init_kernel<<<((size_t)N * 32 + BT - 1) / BT, BT, 0, stream>>>(bufA, dinv, b2, bufB, N);
    agg_scatter_kernel<<<((size_t)E * 32 + BT - 1) / BT, BT, 0, stream>>>(
        e_src, e_dst, dinv, bufA, bufB, E);

    // --- classifier: logits = relu(out2) @ Wc ; log_softmax(logits + bc) ----
    wmma_gemm_kernel<NHID, NCLASS / 16, true><<<gemmGrd, gemmBlk, 0, stream>>>(bufB, wch, wcl, bufA, N);
    logsoftmax_kernel<<<(N + BT - 1) / BT, BT, 0, stream>>>(bufA, bc, out, N);
}
